// CapsuleRouting_23510650978793
// MI455X (gfx1250) — matrix operand order repaired and checked by reading, not compiled, not run on hardware
//
#include <hip/hip_runtime.h>

// ---------------------------------------------------------------------------
// Capsule dynamic routing, fused single kernel for MI455X (gfx1250, wave32).
//
// Roofline: 462 MB input @ 23.3 TB/s => ~20us floor. We hit it by gathering
// each (B=288,C=32,P=16) slice once into the 320KB WGP LDS (fp16 = 288KB,
// a CDNA5-only capability) and running all 3 routing iterations from LDS.
// Reductions run on v_wmma_f32_16x16x32_f16.
// ---------------------------------------------------------------------------

typedef _Float16 v16h __attribute__((ext_vector_type(16)));
typedef _Float16 v8h  __attribute__((ext_vector_type(8)));
typedef float    v8f  __attribute__((ext_vector_type(8)));

#define NB      288
#define NC      32
#define NP      16
#define HW      196
#define NTILES  18          // 288 / 16 B-rows per WMMA tile
#define THREADS 256
#define NWAVES  8
#define EPSF    1e-5f

// ---- LDS layout (bytes) ---------------------------------------------------
#define OFF_U    0
#define SZ_U     (NB * NC * NP * 2)            // 294912: u slice, fp16
#define OFF_VH1  (OFF_U + SZ_U)                // v1 * scale, fp16  (1024 B)
#define OFF_VH2  (OFF_VH1 + 1024)              // v2 * scale, fp16  (1024 B)
#define OFF_CS   (OFF_VH2 + 1024)              // per-wave softmax coeffs: 8*16*32 f32
#define OFF_SS   (OFF_CS + NWAVES * 16 * 32 * 4)  // s accumulator: 32*16 f32
#define OFF_SC   (OFF_SS + NC * NP * 4)        // per-C scale: 32 f32
#define OFF_RED  (OFF_SC + 128)                // min/max reduction scratch: 512 f32
#define SMEM_BYTES (OFF_RED + 2048)            // = 317568 B < 320 KB WGP LDS

// ---------------------------------------------------------------------------
// Agreement via WMMA. For C-half `half` (C in [16*half, 16*half+16)):
//   D[m=B-row, n=C-col] += sum_P u[m, 16*half+n, P] * vh[16*half+n, P]
// realized as 8 chained v_wmma_f32_16x16x32_f16 with K running over (Crel,P).
// A fragment as loaded here (16-bit A 16x32): lane L: m=L%16, h-elem j -> K=j+16*hi
// B fragment as loaded here (16-bit B 32x16): lane L: n=L%16, h-elem j -> K=j+16*hi
//   => per chunk kc, this lane's B rows all belong to Crel = 2*kc + hi, so the
//      block-diagonal mask is a single lane predicate (n == 2*kc + hi).
// ---------------------------------------------------------------------------
__device__ __forceinline__ v8f agr_wmma(const _Float16* __restrict__ uS,
                                        const _Float16* __restrict__ vh,
                                        int t, int half, int hi, int n, v8f D)
{
    const _Float16* uRow  = uS + ((t * 16 + n) * NC + half * 16) * NP; // m = n
    const _Float16* vHalf = vh + half * 16 * NP;
#pragma unroll
    for (int kc = 0; kc < 8; ++kc) {
        // A: this lane's K-elems = one 16-half contiguous run of the u row
        v8h alo = *(const v8h*)(uRow + 32 * kc + 16 * hi);       // Crel=2kc+hi, P=0..7
        v8h ahi = *(const v8h*)(uRow + 32 * kc + 16 * hi + 8);   // Crel=2kc+hi, P=8..15
        v16h A  = __builtin_shufflevector(alo, ahi,
                    0,1,2,3,4,5,6,7,8,9,10,11,12,13,14,15);
        // B: nonzero only for the lane whose column matches Crel = 2kc+hi
        v16h Bm = {};
        if (n == 2 * kc + hi)
            Bm = *(const v16h*)(vHalf + n * NP);                 // v[C=16h+n, 0..15]
        D = __builtin_amdgcn_wmma_f32_16x16x32_f16(false, A, false, Bm,
                                                   (short)0, D, false, false);
    }
    return D;
}

__global__ void __launch_bounds__(THREADS)
caps_route_kernel(const float* __restrict__ u, float* __restrict__ out, int vTot)
{
    extern __shared__ __align__(16) char smem[];
    _Float16* uS  = (_Float16*)(smem + OFF_U);
    _Float16* vh1 = (_Float16*)(smem + OFF_VH1);
    _Float16* vh2 = (_Float16*)(smem + OFF_VH2);
    float*    cS  = (float*)(smem + OFF_CS);
    float*    sS  = (float*)(smem + OFF_SS);
    float*    scS = (float*)(smem + OFF_SC);
    float*    red = (float*)(smem + OFF_RED);

    const int tid  = threadIdx.x;
    const int lane = tid & 31;
    const int wv   = tid >> 5;
    const int hi   = lane >> 4;   // wave32 half
    const int n    = lane & 15;   // WMMA column / row-within-half

    const int bIdx = blockIdx.x / HW;
    const int hw   = blockIdx.x % HW;
    const int locBase = bIdx * (NB * NC * NP * HW) + hw;

    // ---- Phase 0: gather (B,C,P) slice -> LDS fp16 (one HBM pass; lines ----
    // ---- shared across the 32 workgroups of neighboring hw via L2)      ----
#pragma unroll 4
    for (int i = tid; i < NB * NC * NP; i += THREADS)
        uS[i] = (_Float16)u[locBase + i * HW];
    __syncthreads();

    // ---- Phase 1: pose norms, min/max over B, per-C scale = 1/(max-min) ----
    {
        const int c = tid & 31, g = tid >> 5;
        float mx = -1e30f, mn = 1e30f;
        for (int B = g; B < NB; B += 8) {
            const _Float16* p = uS + (B * NC + c) * NP;
            float s = 0.f;
#pragma unroll
            for (int k = 0; k < NP; ++k) { float x = (float)p[k]; s += x * x; }
            float nrm = sqrtf(s + EPSF);
            mx = fmaxf(mx, nrm);
            mn = fminf(mn, nrm);
        }
        red[tid] = mx; red[256 + tid] = mn;
        __syncthreads();
        if (tid < NC) {
            float M = red[tid], m = red[256 + tid];
#pragma unroll
            for (int g2 = 1; g2 < 8; ++g2) {
                M = fmaxf(M, red[g2 * 32 + tid]);
                m = fminf(m, red[256 + g2 * 32 + tid]);
            }
            scS[tid] = 1.0f / (M - m);
        }
        __syncthreads();
    }

    // ---- Phase 2: three routing iterations, entirely from LDS ----
    float* cSw = cS + wv * (16 * 32);   // per-wave coefficient staging

    for (int pass = 1; pass <= 3; ++pass) {
        for (int i = tid; i < NC * NP; i += THREADS) sS[i] = 0.f;
        __syncthreads();

        for (int t = wv; t < NTILES; t += NWAVES) {
            if (pass >= 2) {
                // r[B-tile, C] = sum of agreements with all previous v's
                v8f D0 = {}, D1 = {};
                D0 = agr_wmma(uS, vh1, t, 0, hi, n, D0);
                D1 = agr_wmma(uS, vh1, t, 1, hi, n, D1);
                if (pass == 3) {
                    D0 = agr_wmma(uS, vh2, t, 0, hi, n, D0);
                    D1 = agr_wmma(uS, vh2, t, 1, hi, n, D1);
                }
                // per-row softmax over C: lane holds row j+8*hi, cols n, n+16
#pragma unroll
                for (int j = 0; j < 8; ++j) {
                    float r0 = D0[j], r1 = D1[j];
                    float m = fmaxf(r0, r1);
#pragma unroll
                    for (int off = 1; off < 16; off <<= 1)
                        m = fmaxf(m, __shfl_xor(m, off));
                    float e0 = __expf(r0 - m), e1 = __expf(r1 - m);
                    float s = e0 + e1;
#pragma unroll
                    for (int off = 1; off < 16; off <<= 1)
                        s += __shfl_xor(s, off);
                    float inv = 1.0f / s;
                    int row = j + 8 * hi;
                    cSw[row * 32 + n]      = e0 * inv;
                    cSw[row * 32 + n + 16] = e1 * inv;
                }
                // cross-lane exchange through (per-wave) LDS: drain DS queue
                asm volatile("s_wait_dscnt 0" ::: "memory");
            }

            // s[c, :] += sum_{B in tile} c[B,c] * u[B,c,:]  (one K-padded WMMA)
            // A = coeffs broadcast over M (K rows 0..15 live at j<8, K=j+8*hi)
            // B rows K=0..15 live at j<8, K=j+8*hi
            for (int c = 0; c < NC; ++c) {
                v16h A = {}, Bm = {};
#pragma unroll
                for (int j = 0; j < 8; ++j) {
                    float coef = (pass == 1) ? 0.03125f
                                             : cSw[(j + 8 * hi) * 32 + c];
                    A[j] = (_Float16)coef;
                }
                {
#pragma unroll
                    for (int j = 0; j < 8; ++j)
                        Bm[j] = uS[((t * 16 + j + 8 * hi) * NC + c) * NP + n];
                }
                v8f Ds = {};
                Ds = __builtin_amdgcn_wmma_f32_16x16x32_f16(false, A, false, Bm,
                                                            (short)0, Ds, false, false);
                if (lane < 16)                       // D rows identical; fold scale here
                    atomicAdd(&sS[c * NP + n], Ds[0] * scS[c]);
            }
        }
        __syncthreads();

        // ---- squash; stash v*scale (fp16) for agreement, or emit outputs ----
        if (tid < NC) {
            const int c = tid;
            float sv[NP];
            float nrm2 = EPSF;
#pragma unroll
            for (int p = 0; p < NP; ++p) { sv[p] = sS[c * NP + p]; nrm2 += sv[p] * sv[p]; }
            float inv1p = 1.0f / (1.0f + sqrtf(nrm2));   // squash(s) = s / (1 + |s|)
            if (pass < 3) {
                _Float16* dst = (pass == 1) ? vh1 : vh2;
                float sc = scS[c];
#pragma unroll
                for (int p = 0; p < NP; ++p)
                    dst[c * NP + p] = (_Float16)(sv[p] * inv1p * sc);
            } else {
                float a2 = EPSF;
#pragma unroll
                for (int p = 0; p < NP; ++p) {
                    float vv = sv[p] * inv1p;
                    a2 += vv * vv;
                    out[((bIdx * NC + c) * NP + p) * HW + hw] = vv;
                }
                out[vTot + (bIdx * NC + c) * HW + hw] = sqrtf(a2);
            }
        }
        __syncthreads();
    }
}

extern "C" void kernel_launch(void* const* d_in, const int* in_sizes, int n_in,
                              void* d_out, int out_size, void* d_ws, size_t ws_size,
                              hipStream_t stream)
{
    (void)n_in; (void)d_ws; (void)ws_size; (void)out_size;
    const float* u = (const float*)d_in[0];   // d_in[1] ("a") unused in dynamic mode
    float* out = (float*)d_out;

    const int nb = in_sizes[0] / (NB * NC * NP * HW);   // = 4
    const int vTot = nb * NC * NP * HW;                 // v output elements

    (void)hipFuncSetAttribute((const void*)caps_route_kernel,
                              hipFuncAttributeMaxDynamicSharedMemorySize,
                              SMEM_BYTES);

    dim3 grid(nb * HW);
    caps_route_kernel<<<grid, THREADS, SMEM_BYTES, stream>>>(u, out, vTot);
}
